// QLoRABigNet_62818191671847
// MI455X (gfx1250) — matrix operand order repaired and checked by reading, not compiled, not run on hardware
//
#include <hip/hip_runtime.h>
#include <hip/hip_bf16.h>

typedef __attribute__((ext_vector_type(16))) _Float16 v16h;
typedef __attribute__((ext_vector_type(8)))  _Float16 v8h;
typedef __attribute__((ext_vector_type(4)))  _Float16 v4h;
typedef __attribute__((ext_vector_type(8)))  float    v8f;

#define DIMN    1024
#define TOKENS  16384
#define LORA_R  32
#define GROUPSZ 128

static __device__ __forceinline__ v8f wmma16(v16h a, v16h b, v8f c) {
    // D = A(16x32 f16) * B(32x16 f16) + C(16x16 f32)
    return __builtin_amdgcn_wmma_f32_16x16x32_f16(
        /*neg_a=*/false, a, /*neg_b=*/false, b,
        /*c_mod=*/(short)0, c, /*reuse_a=*/false, /*reuse_b=*/false);
}

// gfx1250 async global->LDS copy (ASYNCcnt-tracked, no VGPR round trip).
// ldsaddr = wave-relative LDS byte address (low 32 bits of generic pointer).
static __device__ __forceinline__ void async_b128(unsigned int ldsaddr,
                                                  unsigned long long gaddr) {
    asm volatile("global_load_async_to_lds_b128 %0, %1, off"
                 :: "v"(ldsaddr), "v"(gaddr) : "memory");
}
#define ASYNC_WAIT(n) asm volatile("s_wait_asynccnt %0" :: "i"(n) : "memory")

// ---------------------------------------------------------------------------
// prep: res32 = x (f32 copy), h16 = f16(x)
// ---------------------------------------------------------------------------
__global__ __launch_bounds__(256) void prep_kernel(const float* __restrict__ x,
                                                   float* __restrict__ res32,
                                                   _Float16* __restrict__ h16) {
    int idx = (blockIdx.x * 256 + threadIdx.x) * 4;
    float4 v = *(const float4*)(x + idx);
    *(float4*)(res32 + idx) = v;
    v4h h;
    h.x = (_Float16)v.x; h.y = (_Float16)v.y;
    h.z = (_Float16)v.z; h.w = (_Float16)v.w;
    *(v4h*)(h16 + idx) = h;
}

// ---------------------------------------------------------------------------
// dequant: W16[o][k] = f16((q/15*2-1) * norm[o][k/128])   (one linear, 1024x1024)
// ---------------------------------------------------------------------------
__global__ __launch_bounds__(256) void dequant_kernel(const int* __restrict__ q,
                                                      const float* __restrict__ nrm,
                                                      _Float16* __restrict__ w16) {
    int idx4 = blockIdx.x * 256 + threadIdx.x;   // 4-elem groups
    int base = idx4 * 4;
    int o = base >> 10;
    int k = base & (DIMN - 1);
    float s = nrm[o * (DIMN / GROUPSZ) + (k >> 7)];
    int4 qv = *(const int4*)(q + base);
    const float c = 2.0f / 15.0f;
    v4h out;
    out.x = (_Float16)(((float)qv.x * c - 1.0f) * s);
    out.y = (_Float16)(((float)qv.y * c - 1.0f) * s);
    out.z = (_Float16)(((float)qv.z * c - 1.0f) * s);
    out.w = (_Float16)(((float)qv.w * c - 1.0f) * s);
    *(v4h*)(w16 + base) = out;
}

// ---------------------------------------------------------------------------
// generic f32 -> f16 (used for lora_b)
// ---------------------------------------------------------------------------
__global__ __launch_bounds__(256) void f32_to_f16_kernel(const float* __restrict__ in,
                                                         _Float16* __restrict__ out) {
    int idx = (blockIdx.x * 256 + threadIdx.x) * 4;
    float4 v = *(const float4*)(in + idx);
    v4h h;
    h.x = (_Float16)v.x; h.y = (_Float16)v.y;
    h.z = (_Float16)v.z; h.w = (_Float16)v.w;
    *(v4h*)(out + idx) = h;
}

// ---------------------------------------------------------------------------
// lora_t: t16[m][r] = f16( sum_k h16[m][k] * la[r][k] ),  r = lane (0..31)
// ---------------------------------------------------------------------------
__global__ __launch_bounds__(256) void lora_t_kernel(const _Float16* __restrict__ h16,
                                                     const float* __restrict__ la,
                                                     _Float16* __restrict__ t16) {
    int wave = threadIdx.x >> 5;
    int lane = threadIdx.x & 31;
    int m = blockIdx.x * 8 + wave;
    const _Float16* hrow = h16 + (size_t)m * DIMN;
    const float* larow = la + (size_t)lane * DIMN;
    float acc = 0.0f;
#pragma unroll 4
    for (int k = 0; k < DIMN; k += 8) {
        v8h hv = *(const v8h*)(hrow + k);
        float4 l0 = *(const float4*)(larow + k);
        float4 l1 = *(const float4*)(larow + k + 4);
        acc += (float)hv[0] * l0.x + (float)hv[1] * l0.y +
               (float)hv[2] * l0.z + (float)hv[3] * l0.w +
               (float)hv[4] * l1.x + (float)hv[5] * l1.y +
               (float)hv[6] * l1.z + (float)hv[7] * l1.w;
    }
    t16[(size_t)m * LORA_R + lane] = (_Float16)acc;
}

// ---------------------------------------------------------------------------
// Main WMMA GEMM:  out[m][n] = sum_k A16[m][k]*W16[n][k] + bias[n]
//                              + sum_r t16[m][r]*lb16[n][r]      (LoRA tail)
//                  (+ resid[m][n] if RES)  (ReLU if RELU)
// BM=128 BN=128 BK=32, 256 threads = 8 waves, wave tile 32x64 (2x4 WMMAs).
// Double-buffered LDS fed by global_load_async_to_lds_b128 (ASYNCcnt).
// ---------------------------------------------------------------------------
#define BM 128
#define BN 128
#define BK 32
#define LDK 40   // padded LDS row stride in halfs (80B, 16B-aligned chunks)

template<bool RELU, bool RES, bool W32>
__global__ __launch_bounds__(256) void gemm_kernel(const _Float16* __restrict__ A,
                                                   const _Float16* __restrict__ W,
                                                   const float* __restrict__ bias,
                                                   const _Float16* __restrict__ T,
                                                   const _Float16* __restrict__ LB,
                                                   const float* __restrict__ resid,
                                                   float* __restrict__ out32,
                                                   _Float16* __restrict__ out16) {
    __shared__ __attribute__((aligned(16))) _Float16 As[2][BM * LDK];
    __shared__ __attribute__((aligned(16))) _Float16 Bs[2][BN * LDK];

    const int tid   = threadIdx.x;
    const int lane  = tid & 31;
    const int wave  = tid >> 5;
    const int waveM = wave & 3;    // 4 waves along M (32 rows each)
    const int waveN = wave >> 2;   // 2 waves along N (64 cols each)
    const int hs    = lane >> 4;   // half-wave select
    const int l16   = lane & 15;

    const int tileN = blockIdx.x * BN;
    const int tileM = blockIdx.y * BM;

    // this thread's two staging chunks (row, 16B segment) — same for A and B
    const int c0row = tid >> 2,          c0seg = tid & 3;
    const int c1row = (tid + 256) >> 2,  c1seg = (tid + 256) & 3;

    v8f acc[2][4];
#pragma unroll
    for (int sm = 0; sm < 2; ++sm)
#pragma unroll
        for (int sn = 0; sn < 4; ++sn)
            acc[sm][sn] = {};

    // issue stage 0 into buffer 0
    {
        async_b128((unsigned int)(size_t)&As[0][c0row * LDK + c0seg * 8],
                   (unsigned long long)(size_t)&A[(size_t)(tileM + c0row) * DIMN + c0seg * 8]);
        async_b128((unsigned int)(size_t)&Bs[0][c0row * LDK + c0seg * 8],
                   (unsigned long long)(size_t)&W[(size_t)(tileN + c0row) * DIMN + c0seg * 8]);
        async_b128((unsigned int)(size_t)&As[0][c1row * LDK + c1seg * 8],
                   (unsigned long long)(size_t)&A[(size_t)(tileM + c1row) * DIMN + c1seg * 8]);
        async_b128((unsigned int)(size_t)&Bs[0][c1row * LDK + c1seg * 8],
                   (unsigned long long)(size_t)&W[(size_t)(tileN + c1row) * DIMN + c1seg * 8]);
    }

    for (int kk = 0; kk < DIMN; kk += BK) {
        const int buf = (kk >> 5) & 1;
        const int nxt = buf ^ 1;
        if (kk + BK < DIMN) {
            const int kn = kk + BK;
            // prefetch next K-slice into the other buffer while we compute
            async_b128((unsigned int)(size_t)&As[nxt][c0row * LDK + c0seg * 8],
                       (unsigned long long)(size_t)&A[(size_t)(tileM + c0row) * DIMN + kn + c0seg * 8]);
            async_b128((unsigned int)(size_t)&Bs[nxt][c0row * LDK + c0seg * 8],
                       (unsigned long long)(size_t)&W[(size_t)(tileN + c0row) * DIMN + kn + c0seg * 8]);
            async_b128((unsigned int)(size_t)&As[nxt][c1row * LDK + c1seg * 8],
                       (unsigned long long)(size_t)&A[(size_t)(tileM + c1row) * DIMN + kn + c1seg * 8]);
            async_b128((unsigned int)(size_t)&Bs[nxt][c1row * LDK + c1seg * 8],
                       (unsigned long long)(size_t)&W[(size_t)(tileN + c1row) * DIMN + kn + c1seg * 8]);
            ASYNC_WAIT(4);   // in-order completion: current buffer's 4 are done
        } else {
            ASYNC_WAIT(0);
        }
        __syncthreads();     // all waves' current-slice data visible in LDS

        // A fragments: lane<16 holds K0..7 & 16..23; lane>=16 holds K8..15 & 24..31
        v16h af[2];
#pragma unroll
        for (int sm = 0; sm < 2; ++sm) {
            int r = (waveM * 32 + sm * 16 + l16) * LDK;
            v8h lo = *(const v8h*)&As[buf][r + hs * 8];
            v8h hi = *(const v8h*)&As[buf][r + 16 + hs * 8];
            af[sm] = __builtin_shufflevector(lo, hi, 0,1,2,3,4,5,6,7,8,9,10,11,12,13,14,15);
        }
        // B fragments: lane = column; lane<16 holds K0..7 & 16..23; lane>=16 holds K8..15 & 24..31
        v16h bf[4];
#pragma unroll
        for (int sn = 0; sn < 4; ++sn) {
            int r = (waveN * 64 + sn * 16 + l16) * LDK;
            v8h lo = *(const v8h*)&Bs[buf][r + hs * 8];
            v8h hi = *(const v8h*)&Bs[buf][r + 16 + hs * 8];
            bf[sn] = __builtin_shufflevector(lo, hi, 0,1,2,3,4,5,6,7,8,9,10,11,12,13,14,15);
        }
#pragma unroll
        for (int sm = 0; sm < 2; ++sm)
#pragma unroll
            for (int sn = 0; sn < 4; ++sn)
                acc[sm][sn] = wmma16(af[sm], bf[sn], acc[sm][sn]);

        __syncthreads();     // everyone done reading buf before it is re-filled
    }

    // ---- LoRA tail: one extra K=32 WMMA with A=t16, B=lb16 (global, rows are 64B)
    {
        v16h af[2];
#pragma unroll
        for (int sm = 0; sm < 2; ++sm) {
            size_t mg = (size_t)(tileM + waveM * 32 + sm * 16 + l16) * LORA_R;
            v8h lo = *(const v8h*)&T[mg + hs * 8];
            v8h hi = *(const v8h*)&T[mg + 16 + hs * 8];
            af[sm] = __builtin_shufflevector(lo, hi, 0,1,2,3,4,5,6,7,8,9,10,11,12,13,14,15);
        }
        v16h bf[4];
#pragma unroll
        for (int sn = 0; sn < 4; ++sn) {
            size_t ng = (size_t)(tileN + waveN * 64 + sn * 16 + l16) * LORA_R;
            v8h lo = *(const v8h*)&LB[ng + hs * 8];
            v8h hi = *(const v8h*)&LB[ng + 16 + hs * 8];
            bf[sn] = __builtin_shufflevector(lo, hi, 0,1,2,3,4,5,6,7,8,9,10,11,12,13,14,15);
        }
#pragma unroll
        for (int sm = 0; sm < 2; ++sm)
#pragma unroll
            for (int sn = 0; sn < 4; ++sn)
                acc[sm][sn] = wmma16(af[sm], bf[sn], acc[sm][sn]);
    }

    // ---- epilogue: D layout -> lane holds col l16, rows hs*8 .. hs*8+7
#pragma unroll
    for (int sm = 0; sm < 2; ++sm) {
#pragma unroll
        for (int sn = 0; sn < 4; ++sn) {
            int n = tileN + waveN * 64 + sn * 16 + l16;
            float bv = bias[n];
            int mb = tileM + waveM * 32 + sm * 16 + hs * 8;
#pragma unroll
            for (int r = 0; r < 8; ++r) {
                int m = mb + r;
                float v = acc[sm][sn][r] + bv;
                if (RES)  v += resid[(size_t)m * DIMN + n];
                if (RELU) v = fmaxf(v, 0.0f);
                if (W32)  out32[(size_t)m * DIMN + n] = v;
                out16[(size_t)m * DIMN + n] = (_Float16)v;
            }
        }
    }
}

// ---------------------------------------------------------------------------
// LayerNorm over each 1024-wide row; writes f32 (next residual) and f16.
// ---------------------------------------------------------------------------
__global__ __launch_bounds__(256) void ln_kernel(const float* __restrict__ in,
                                                 const float* __restrict__ gamma,
                                                 const float* __restrict__ beta,
                                                 float* __restrict__ act32,
                                                 _Float16* __restrict__ act16) {
    __shared__ float s1[256];
    __shared__ float s2[256];
    int row = blockIdx.x;
    int tid = threadIdx.x;
    float4 v = ((const float4*)(in + (size_t)row * DIMN))[tid];
    s1[tid] = v.x + v.y + v.z + v.w;
    s2[tid] = v.x * v.x + v.y * v.y + v.z * v.z + v.w * v.w;
    __syncthreads();
    for (int off = 128; off > 0; off >>= 1) {
        if (tid < off) { s1[tid] += s1[tid + off]; s2[tid] += s2[tid + off]; }
        __syncthreads();
    }
    float mu  = s1[0] * (1.0f / DIMN);
    float var = s2[0] * (1.0f / DIMN) - mu * mu;
    float inv = rsqrtf(var + 1e-5f);
    float4 g = ((const float4*)gamma)[tid];
    float4 b = ((const float4*)beta)[tid];
    float4 o;
    o.x = (v.x - mu) * inv * g.x + b.x;
    o.y = (v.y - mu) * inv * g.y + b.y;
    o.z = (v.z - mu) * inv * g.z + b.z;
    o.w = (v.w - mu) * inv * g.w + b.w;
    ((float4*)(act32 + (size_t)row * DIMN))[tid] = o;
    v4h h;
    h.x = (_Float16)o.x; h.y = (_Float16)o.y;
    h.z = (_Float16)o.z; h.w = (_Float16)o.w;
    ((v4h*)(act16 + (size_t)row * DIMN))[tid] = h;
}

// ---------------------------------------------------------------------------
// host launcher
// ---------------------------------------------------------------------------
extern "C" void kernel_launch(void* const* d_in, const int* in_sizes, int n_in,
                              void* d_out, int out_size, void* d_ws, size_t ws_size,
                              hipStream_t stream) {
    (void)in_sizes; (void)n_in; (void)out_size; (void)ws_size;
    const float* x     = (const float*)d_in[0];
    const int*   q     = (const int*)  d_in[1];
    const float* nrm   = (const float*)d_in[2];
    const float* bias  = (const float*)d_in[3];
    const float* la    = (const float*)d_in[4];
    const float* lb    = (const float*)d_in[5];
    const float* lng   = (const float*)d_in[6];
    const float* lnb   = (const float*)d_in[7];
    float* out = (float*)d_out;

    // workspace layout (bytes)
    char* base = (char*)d_ws;
    _Float16* h16a = (_Float16*)(base);                          // 33,554,432
    _Float16* h16b = (_Float16*)(base + 33554432ull);            // 33,554,432
    _Float16* t16  = (_Float16*)(base + 67108864ull);            //  1,048,576
    _Float16* W16  = (_Float16*)(base + 68157440ull);            //  2,097,152
    _Float16* lb16 = (_Float16*)(base + 70254592ull);            //     65,536
    float*    o32  = (float*)   (base + 70320128ull);            // 67,108,864
    float*    res32= (float*)   (base + 137428992ull);           // 67,108,864

    prep_kernel<<<TOKENS * DIMN / 1024, 256, 0, stream>>>(x, res32, h16a);

    _Float16* hin = h16a;
    _Float16* hout = h16b;
    for (int blk = 0; blk < 6; ++blk) {
        for (int j = 0; j < 3; ++j) {
            int i = 3 * blk + j;
            dequant_kernel<<<DIMN * DIMN / 1024, 256, 0, stream>>>(
                q + (size_t)i * DIMN * DIMN,
                nrm + (size_t)i * DIMN * (DIMN / GROUPSZ), W16);
            f32_to_f16_kernel<<<DIMN * LORA_R / 1024, 256, 0, stream>>>(
                lb + (size_t)i * DIMN * LORA_R, lb16);
            lora_t_kernel<<<TOKENS / 8, 256, 0, stream>>>(
                hin, la + (size_t)i * LORA_R * DIMN, t16);

            dim3 grid(DIMN / BN, TOKENS / BM);
            if (j < 2) {
                gemm_kernel<true, false, false><<<grid, 256, 0, stream>>>(
                    hin, W16, bias + (size_t)i * DIMN, t16, lb16,
                    nullptr, nullptr, hout);
            } else {
                float* dst32 = (blk == 5) ? out : o32;
                gemm_kernel<false, true, true><<<grid, 256, 0, stream>>>(
                    hin, W16, bias + (size_t)i * DIMN, t16, lb16,
                    res32, dst32, hout);
            }
            _Float16* tmp = hin; hin = hout; hout = tmp;
        }
        if (blk < 5) {
            ln_kernel<<<TOKENS, 256, 0, stream>>>(
                o32, lng + (size_t)blk * DIMN, lnb + (size_t)blk * DIMN,
                res32, hin);
        }
    }
}
